// LoraLinear_58918361366727
// MI455X (gfx1250) — matrix operand order repaired and checked by reading, not compiled, not run on hardware
//
#include <hip/hip_runtime.h>
#include <stdint.h>

// Problem constants (match reference)
#define D_IN   4096
#define D_OUT  4096
#define BS     16384   // B*S rows
#define RANK   16
#define NADAPT 8
#define SEQ    2048

typedef __bf16 v16bf __attribute__((ext_vector_type(16)));
typedef float  v8f   __attribute__((ext_vector_type(8)));
typedef int    v4i   __attribute__((ext_vector_type(4)));

#if __has_builtin(__builtin_amdgcn_global_load_async_to_lds_b128)
#define HAVE_ASYNC 1
#endif

// round-to-nearest-even fp32 -> bf16 (bit pattern as ushort)
static __device__ __forceinline__ unsigned short f2bf(float f) {
  union { float f; unsigned u; } c; c.f = f;
  unsigned u = c.u;
  return (unsigned short)((u + 0x7fffu + ((u >> 16) & 1u)) >> 16);
}

// ---------------------------------------------------------------------------
// Kernel 1: W fp32 -> bf16 (single streaming pass; W_bf16 = 33.5 MB, fits L2)
// ---------------------------------------------------------------------------
__global__ __launch_bounds__(256) void k_cvt_w(const float* __restrict__ W,
                                               unsigned short* __restrict__ wb) {
  const size_t n4 = (size_t)D_OUT * D_IN / 4;
  for (size_t i = blockIdx.x * blockDim.x + threadIdx.x; i < n4;
       i += (size_t)gridDim.x * blockDim.x) {
    float4 v = ((const float4*)W)[i];
    ushort4 o;
    o.x = f2bf(v.x); o.y = f2bf(v.y); o.z = f2bf(v.z); o.w = f2bf(v.w);
    ((ushort4*)wb)[i] = o;
  }
}

// ---------------------------------------------------------------------------
// Kernel 2: lora_b [A][D_OUT][16] fp32 -> bf16 rank-padded to K=32 (zeros)
// ---------------------------------------------------------------------------
__global__ __launch_bounds__(256) void k_build_lb(const float* __restrict__ lbsrc,
                                                  unsigned short* __restrict__ lb) {
  const size_t n = (size_t)NADAPT * D_OUT * 32;
  for (size_t i = blockIdx.x * blockDim.x + threadIdx.x; i < n;
       i += (size_t)gridDim.x * blockDim.x) {
    size_t a = i / ((size_t)D_OUT * 32);
    size_t r = i % ((size_t)D_OUT * 32);
    size_t o = r / 32, k = r % 32;
    lb[i] = (k < RANK) ? f2bf(lbsrc[(a * D_OUT + o) * RANK + k]) : (unsigned short)0;
  }
}

// ---------------------------------------------------------------------------
// Kernel 3: one streaming pass over x: emit x_bf16 AND inter = x @ a[adapter].T
// (inter stored bf16, rank-padded to 32).  One block per row, 256 threads.
// ---------------------------------------------------------------------------
__global__ __launch_bounds__(256) void k_rows(const float* __restrict__ x,
                                              const float* __restrict__ la,
                                              const int* __restrict__ aidx,
                                              unsigned short* __restrict__ xb,
                                              unsigned short* __restrict__ ib) {
  const int row = blockIdx.x;
  const int adapter = aidx[row / SEQ];
  const float* xr = x + (size_t)row * D_IN;
  const float* ar = la + (size_t)adapter * RANK * D_IN;
  const int t = threadIdx.x;
  const int lane = t & 31, w = t >> 5;
  const int base = t * 16;   // 256 threads * 16 elems = 4096

  float acc[RANK];
#pragma unroll
  for (int r = 0; r < RANK; ++r) acc[r] = 0.0f;

  union { unsigned short s[16]; uint4 q[2]; } pk;
#pragma unroll
  for (int e = 0; e < 16; ++e) {
    float xv = xr[base + e];
    pk.s[e] = f2bf(xv);
#pragma unroll
    for (int r = 0; r < RANK; ++r) acc[r] += xv * ar[r * D_IN + base + e];
  }
  uint4* xdst = (uint4*)(xb + (size_t)row * D_IN + base);
  xdst[0] = pk.q[0];
  xdst[1] = pk.q[1];

  // wave32 shuffle reduction, then cross-wave via LDS
#pragma unroll
  for (int r = 0; r < RANK; ++r) {
#pragma unroll
    for (int off = 16; off > 0; off >>= 1)
      acc[r] += __shfl_down(acc[r], off, 32);
  }
  __shared__ float red[8][RANK];
  if (lane == 0) {
#pragma unroll
    for (int r = 0; r < RANK; ++r) red[w][r] = acc[r];
  }
  __syncthreads();
  if (t < RANK) {
    float s = 0.0f;
#pragma unroll
    for (int ww = 0; ww < 8; ++ww) s += red[ww][t];
    ib[(size_t)row * 32 + t] = f2bf(s);
  } else if (t < 32) {
    ib[(size_t)row * 32 + t] = 0;   // rank padding
  }
}

// ---------------------------------------------------------------------------
// Kernel 4: fused GEMM.  128x128 tile/block, 8 waves, wave = 32x64 (2x4 WMMA).
// K loop: 128 tiles of 32 from x_bf16/W_bf16 + 1 fused LoRA tile (inter/lora_b).
// B tiles double-buffered in LDS via async-to-LDS; A fragments loaded straight
// from global in WMMA register layout.
// ---------------------------------------------------------------------------
#define LDS_BSTRIDE 40   // 32 bf16 + 8 pad => 80B rows, 16B aligned, bank-spread

union AFrag { v16bf v; uint4 q[2]; };

static __device__ __forceinline__ void cp16(unsigned short* l, const unsigned short* g) {
#ifdef HAVE_ASYNC
  __builtin_amdgcn_global_load_async_to_lds_b128(
      (__attribute__((address_space(1))) v4i*)(v4i*)g,
      (__attribute__((address_space(3))) v4i*)(v4i*)l,
      0, 0);
#else
  *(uint4*)l = *(const uint4*)g;
#endif
}

static __device__ __forceinline__ void wait_async_le2() {
#ifdef HAVE_ASYNC
#if __has_builtin(__builtin_amdgcn_s_wait_asynccnt)
  __builtin_amdgcn_s_wait_asynccnt(2);
#else
  asm volatile("s_wait_asynccnt 0x2" ::: "memory");
#endif
#endif
}
static __device__ __forceinline__ void wait_async_le0() {
#ifdef HAVE_ASYNC
#if __has_builtin(__builtin_amdgcn_s_wait_asynccnt)
  __builtin_amdgcn_s_wait_asynccnt(0);
#else
  asm volatile("s_wait_asynccnt 0x0" ::: "memory");
#endif
#endif
}

__global__ __launch_bounds__(256) void k_gemm(const unsigned short* __restrict__ xb,
                                              const unsigned short* __restrict__ wb,
                                              const unsigned short* __restrict__ ib,
                                              const unsigned short* __restrict__ lb,
                                              const float* __restrict__ bias,
                                              const int* __restrict__ aidx,
                                              float* __restrict__ out) {
  __shared__ unsigned short ldsB[2][128 * LDS_BSTRIDE];

  const int t    = threadIdx.x;
  const int lane = t & 31, w = t >> 5;
  const int l    = lane & 15, hi = lane >> 4;
  const int wm   = w & 3,  wn = w >> 2;            // 4 (M) x 2 (N) waves
  const int row0 = blockIdx.y * 128;
  const int col0 = blockIdx.x * 128;
  const int adapter = aidx[row0 / SEQ];
  const int NK = 129;                               // 128 main + 1 LoRA K-tile

  v8f acc[2][4];
#pragma unroll
  for (int m = 0; m < 2; ++m)
#pragma unroll
    for (int n = 0; n < 4; ++n)
#pragma unroll
      for (int r = 0; r < 8; ++r) acc[m][n][r] = 0.0f;

  // --- issue B-tile copy (128 N rows x 32 K bf16 = 512 x16B chunks) ---------
  auto issueB = [&](int kt, int buf) {
    const unsigned short* src;
    size_t stride; int kk;
    if (kt < 128) { src = wb;                                stride = D_IN; kk = kt * 32; }
    else          { src = lb + (size_t)adapter * D_OUT * 32; stride = 32;   kk = 0;      }
#pragma unroll
    for (int i = 0; i < 2; ++i) {
      int idx = t + 256 * i;
      int n = idx >> 2, c = idx & 3;
      cp16(&ldsB[buf][n * LDS_BSTRIDE + c * 8],
           src + (size_t)(col0 + n) * stride + kk + c * 8);
    }
  };

  issueB(0, 0);

  for (int kt = 0; kt < NK; ++kt) {
    const int buf = kt & 1;
    if (kt + 1 < NK) { issueB(kt + 1, buf ^ 1); wait_async_le2(); }
    else             { wait_async_le0(); }
    __syncthreads();

    // A fragments straight from global in 16-bit A layout:
    // lane l   : row M=l, K {0..7} and {16..23};  lane l+16: K {8..15},{24..31}
    AFrag af[2];
    {
      const unsigned short* asrc;
      size_t stride; int kk;
      if (kt < 128) { asrc = xb; stride = D_IN; kk = kt * 32; }
      else          { asrc = ib; stride = 32;   kk = 0;       }
#pragma unroll
      for (int m = 0; m < 2; ++m) {
        const int row = row0 + wm * 32 + m * 16 + l;
        const unsigned short* p = asrc + (size_t)row * stride + kk + hi * 8;
        af[m].q[0] = *(const uint4*)p;
        af[m].q[1] = *(const uint4*)(p + 16);
        if (kt < 127) __builtin_prefetch(p + 32, 0, 0);   // next K-tile of A
      }
    }

    // B fragments from LDS: lane n holds column n, K {0..7},{16..23} (lo half) / {8..15},{24..31}
    AFrag bf[4];
#pragma unroll
    for (int n = 0; n < 4; ++n) {
      const unsigned short* p = &ldsB[buf][(wn * 64 + n * 16 + l) * LDS_BSTRIDE + hi * 8];
      bf[n].q[0] = *(const uint4*)p;
      bf[n].q[1] = *(const uint4*)(p + 16);
    }

#pragma unroll
    for (int m = 0; m < 2; ++m)
#pragma unroll
      for (int n = 0; n < 4; ++n)
        acc[m][n] = __builtin_amdgcn_wmma_f32_16x16x32_bf16(
            false, af[m].v, false, bf[n].v, (short)0, acc[m][n], false, false);

    __syncthreads();
  }

  // ---- epilogue: + bias, fp32 store.  C layout: VGPR r -> M=r (+8 for hi) --
  float bv[4];
#pragma unroll
  for (int n = 0; n < 4; ++n) bv[n] = bias[col0 + wn * 64 + n * 16 + l];

#pragma unroll
  for (int m = 0; m < 2; ++m)
#pragma unroll
    for (int n = 0; n < 4; ++n) {
      const int col = col0 + wn * 64 + n * 16 + l;
#pragma unroll
      for (int r = 0; r < 8; ++r) {
        const int row = row0 + wm * 32 + m * 16 + hi * 8 + r;
        out[(size_t)row * D_OUT + col] = acc[m][n][r] + bv[n];
      }
    }
}

// ---------------------------------------------------------------------------
extern "C" void kernel_launch(void* const* d_in, const int* in_sizes, int n_in,
                              void* d_out, int out_size, void* d_ws, size_t ws_size,
                              hipStream_t stream) {
  const float* x     = (const float*)d_in[0];
  const float* W     = (const float*)d_in[1];
  const float* bias  = (const float*)d_in[2];
  const float* la    = (const float*)d_in[3];
  const float* lbsrc = (const float*)d_in[4];
  const int*   aidx  = (const int*)d_in[5];
  float* out = (float*)d_out;

  char* ws = (char*)d_ws;
  unsigned short* xb = (unsigned short*)ws;                       // 134,217,728 B
  unsigned short* wb = (unsigned short*)(ws + (size_t)BS * D_IN * 2);
  unsigned short* ib = (unsigned short*)((char*)wb + (size_t)D_OUT * D_IN * 2); // 1 MB
  unsigned short* lb = (unsigned short*)((char*)ib + (size_t)BS * 32 * 2);      // 2 MB

  k_cvt_w   <<<4096, 256, 0, stream>>>(W, wb);
  k_build_lb<<<1024, 256, 0, stream>>>(lbsrc, lb);
  k_rows    <<<BS,   256, 0, stream>>>(x, la, aidx, xb, ib);
  k_gemm    <<<dim3(D_OUT / 128, BS / 128), 256, 0, stream>>>(xb, wb, ib, lb, bias, aidx, out);
}
